// AttnBlock_6760278524340
// MI455X (gfx1250) — matrix operand order repaired and checked by reading, not compiled, not run on hardware
//
#include <hip/hip_runtime.h>

#define B_ 4
#define C_ 512
#define N_ 4096
#define IT 2  // 16-query subtiles per attention block (32 queries/block)

#if defined(__HIP_DEVICE_COMPILE__) && __has_builtin(__builtin_amdgcn_global_load_async_to_lds_b128)
#define ASYNC_LDS 1
#else
#define ASYNC_LDS 0
#endif

typedef __attribute__((ext_vector_type(16))) __bf16 bf16x16;
typedef __attribute__((ext_vector_type(8)))  float  floatx8;
typedef int v4i __attribute__((vector_size(16)));
typedef __attribute__((address_space(1))) v4i* gas_v4i_p;
typedef __attribute__((address_space(3))) v4i* las_v4i_p;

union Frag {
  bf16x16 v;
  uint4 q[2];
  unsigned short u[16];
};

union Pack8 {
  unsigned short u[8];
  uint4 q;
};

__device__ __forceinline__ unsigned short f2bf(float f) {
  union { float f; unsigned u; } c;
  c.f = f;
  unsigned r = c.u + 0x7FFFu + ((c.u >> 16) & 1u);  // round-to-nearest-even
  return (unsigned short)(r >> 16);
}

__device__ __forceinline__ floatx8 wmma_bf16(const Frag& a, const Frag& b, floatx8 c) {
  return __builtin_amdgcn_wmma_f32_16x16x32_bf16(false, a.v, false, b.v, (short)0, c,
                                                 false, false);
}

#if ASYNC_LDS
__device__ __forceinline__ void async_copy_b128(const void* gsrc, void* ldst) {
  auto gv = (__attribute__((address_space(1))) void*)(void*)gsrc;
  auto lv = (__attribute__((address_space(3))) void*)ldst;
  __builtin_amdgcn_global_load_async_to_lds_b128((gas_v4i_p)gv, (las_v4i_p)lv, 0, 0);
}
#endif

// ---------------------------------------------------------------------------
// K0: f32 -> bf16 weight pack
// ---------------------------------------------------------------------------
__global__ __launch_bounds__(256) void cvt_bf16(const float* __restrict__ src,
                                                unsigned short* __restrict__ dst, int n) {
  int i = blockIdx.x * 256 + threadIdx.x;
  if (i < n) dst[i] = f2bf(src[i]);
}

// ---------------------------------------------------------------------------
// K1: GroupNorm, writes h transposed [B][N][C] bf16
// ---------------------------------------------------------------------------
__global__ __launch_bounds__(256) void gn_kernel(const float* __restrict__ x,
                                                 const float* __restrict__ gamma,
                                                 const float* __restrict__ beta,
                                                 unsigned short* __restrict__ hT) {
  const int bb = blockIdx.x >> 5;
  const int gg = blockIdx.x & 31;
  const float* xg = x + ((size_t)bb * C_ + gg * 16) * N_;

  float s = 0.f, ss = 0.f;
  for (int i = threadIdx.x; i < 16 * N_; i += 256) {
    float v = xg[i];
    s += v;
    ss += v * v;
  }
  __shared__ float rs[256], rq[256];
  rs[threadIdx.x] = s;
  rq[threadIdx.x] = ss;
  __syncthreads();
  for (int o = 128; o > 0; o >>= 1) {
    if (threadIdx.x < o) {
      rs[threadIdx.x] += rs[threadIdx.x + o];
      rq[threadIdx.x] += rq[threadIdx.x + o];
    }
    __syncthreads();
  }
  const float cnt = 16.f * N_;
  const float mu = rs[0] / cnt;
  const float var = rq[0] / cnt - mu * mu;
  const float rstd = rsqrtf(var + 1e-6f);

  for (int i = threadIdx.x; i < 16 * N_; i += 256) {
    const int cl = i >> 12;
    const int n = i & (N_ - 1);
    const int c = gg * 16 + cl;
    const float v = (xg[i] - mu) * rstd * gamma[c] + beta[c];
    hT[((size_t)bb * N_ + n) * C_ + c] = f2bf(v);
  }
}

// ---------------------------------------------------------------------------
// K2: fused QKV GEMM.  W stacked [1536][512] bf16, hT [B][N][C] bf16.
// q,k stored transposed [B][N][C]; v stored [B][C][N].
// ---------------------------------------------------------------------------
__global__ __launch_bounds__(256) void qkv_gemm(
    const unsigned short* __restrict__ W, const unsigned short* __restrict__ hT,
    const float* __restrict__ qb, const float* __restrict__ kb,
    const float* __restrict__ vb, unsigned short* __restrict__ qT,
    unsigned short* __restrict__ kT, unsigned short* __restrict__ vbuf) {
  const int wave = threadIdx.x >> 5, lane = threadIdx.x & 31;
  const int ln = lane & 15, lh = lane >> 4;
  const int o0 = blockIdx.x * 128 + (wave >> 2) * 64;
  const int n0 = blockIdx.y * 64 + (wave & 3) * 16;
  const int bb = blockIdx.z;

  floatx8 acc[4] = {};
  const unsigned short* bptr = hT + ((size_t)bb * N_ + n0 + ln) * C_ + 8 * lh;
  const unsigned short* aptr = W + (size_t)(o0 + ln) * C_ + 8 * lh;

  for (int kc = 0; kc < C_; kc += 32) {
    __builtin_prefetch(bptr + kc + 64, 0, 1);
    Frag bfr;
    bfr.q[0] = *(const uint4*)(bptr + kc);
    bfr.q[1] = *(const uint4*)(bptr + kc + 16);
#pragma unroll
    for (int t = 0; t < 4; ++t) {
      Frag afr;
      const unsigned short* ap = aptr + (size_t)(16 * t) * C_ + kc;
      afr.q[0] = *(const uint4*)(ap);
      afr.q[1] = *(const uint4*)(ap + 16);
      acc[t] = wmma_bf16(afr, bfr, acc[t]);
    }
  }

  const int mat = o0 >> 9;  // 0=q 1=k 2=v
  const int ro0 = o0 & 511;
  const float* bias = (mat == 0) ? qb : (mat == 1) ? kb : vb;
  const int n = n0 + ln;
#pragma unroll
  for (int t = 0; t < 4; ++t) {
    const int rb = ro0 + 16 * t + 8 * lh;
    Pack8 pk;
#pragma unroll
    for (int r = 0; r < 8; ++r) pk.u[r] = f2bf(acc[t][r] + bias[rb + r]);
    if (mat < 2) {
      unsigned short* dst = (mat == 0) ? qT : kT;
      *(uint4*)(dst + ((size_t)bb * N_ + n) * C_ + rb) = pk.q;
    } else {
#pragma unroll
      for (int r = 0; r < 8; ++r)
        vbuf[((size_t)bb * C_ + rb + r) * N_ + n] = pk.u[r];
    }
  }
}

// ---------------------------------------------------------------------------
// K3: flash attention, 32 queries per block.  8 waves:
//  - scores: wave w owns keys [J+16w, J+16w+16) vs 32 queries (2 tiles)
//  - output: wave w owns channels [64w, 64w+64) x 32 queries (8 tiles)
// Q tile staged to LDS via GLOBAL_LOAD_ASYNC_TO_LDS when available.
// ---------------------------------------------------------------------------
__global__ __launch_bounds__(256) void attn_kernel(
    const unsigned short* __restrict__ qT, const unsigned short* __restrict__ kT,
    const unsigned short* __restrict__ vbuf, unsigned short* __restrict__ attT) {
  __shared__ __align__(16) unsigned short Qs[16 * IT * C_];   // 32 KB
  __shared__ __align__(16) unsigned short Ps[16 * IT * 128];  // 8 KB
  __shared__ float mpart[8][16 * IT], lpart[8][16 * IT];
  __shared__ float mrun[16 * IT], lrun[16 * IT], mnew[16 * IT], corr[16 * IT];

  const int bb = blockIdx.y;
  const int i0 = blockIdx.x * (16 * IT);
  const int wave = threadIdx.x >> 5, lane = threadIdx.x & 31;
  const int ln = lane & 15, lh = lane >> 4;

  const uint4* qsrc = (const uint4*)(qT + ((size_t)bb * N_ + i0) * C_);
#if ASYNC_LDS
  for (int i = threadIdx.x; i < 16 * IT * C_ / 8; i += 256)
    async_copy_b128(qsrc + i, &Qs[i * 8]);
#if __has_builtin(__builtin_amdgcn_s_wait_asynccnt)
  __builtin_amdgcn_s_wait_asynccnt(0);
#else
  asm volatile("s_wait_asynccnt 0x0" ::: "memory");
#endif
#else
  for (int i = threadIdx.x; i < 16 * IT * C_ / 8; i += 256) ((uint4*)Qs)[i] = qsrc[i];
#endif
  if (threadIdx.x < 16 * IT) {
    mrun[threadIdx.x] = -1e30f;
    lrun[threadIdx.x] = 0.f;
  }
  __syncthreads();

  floatx8 oacc[4][IT] = {};
  const float scale = 0.04419417382415922f;  // 512^-0.5

  for (int J = 0; J < N_; J += 128) {
    // ---- S = (Q^T K) * scale: this wave's 16 keys vs 32 queries ------------
    const int j0 = J + wave * 16;
    floatx8 s[IT] = {};
    const unsigned short* kbp = kT + ((size_t)bb * N_ + j0 + ln) * C_ + 8 * lh;
    for (int kc = 0; kc < C_; kc += 32) {
      __builtin_prefetch(kbp + kc + 64, 0, 1);
      Frag bfr;
      bfr.q[0] = *(const uint4*)(kbp + kc);
      bfr.q[1] = *(const uint4*)(kbp + kc + 16);
#pragma unroll
      for (int it = 0; it < IT; ++it) {
        Frag afr;
        afr.q[0] = *(const uint4*)&Qs[(it * 16 + ln) * C_ + 8 * lh + kc];
        afr.q[1] = *(const uint4*)&Qs[(it * 16 + ln) * C_ + 8 * lh + kc + 16];
        s[it] = wmma_bf16(afr, bfr, s[it]);
      }
    }

    // ---- per-wave row maxima -> LDS ---------------------------------------
#pragma unroll
    for (int it = 0; it < IT; ++it) {
#pragma unroll
      for (int r = 0; r < 8; ++r) {
        float t = s[it][r] * scale;
        s[it][r] = t;
        for (int d = 1; d < 16; d <<= 1) t = fmaxf(t, __shfl_xor(t, d));
        if (ln == 0) mpart[wave][it * 16 + r + 8 * lh] = t;
      }
    }
    __syncthreads();

    if (threadIdx.x < 16 * IT) {
      float m = mrun[threadIdx.x];
#pragma unroll
      for (int w = 0; w < 8; ++w) m = fmaxf(m, mpart[w][threadIdx.x]);
      corr[threadIdx.x] = __expf(mrun[threadIdx.x] - m);
      mnew[threadIdx.x] = m;
      mrun[threadIdx.x] = m;
    }
    __syncthreads();

    // ---- P = exp(S - m_new); row sums; stash P tile in LDS [i][j] ---------
#pragma unroll
    for (int it = 0; it < IT; ++it) {
#pragma unroll
      for (int r = 0; r < 8; ++r) {
        const int row = it * 16 + r + 8 * lh;
        float p = __expf(s[it][r] - mnew[row]);
        Ps[row * 128 + wave * 16 + ln] = f2bf(p);
        for (int d = 1; d < 16; d <<= 1) p += __shfl_xor(p, d);
        if (ln == 0) lpart[wave][row] = p;
      }
    }

    // ---- rescale running output by exp(m_old - m_new) ---------------------
#pragma unroll
    for (int it = 0; it < IT; ++it) {
      const float cf = corr[it * 16 + ln];
#pragma unroll
      for (int ct = 0; ct < 4; ++ct)
#pragma unroll
        for (int r = 0; r < 8; ++r) oacc[ct][it][r] *= cf;
    }
    __syncthreads();

    if (threadIdx.x < 16 * IT) {
      float l = lrun[threadIdx.x] * corr[threadIdx.x];
#pragma unroll
      for (int w = 0; w < 8; ++w) l += lpart[w][threadIdx.x];
      lrun[threadIdx.x] = l;
    }

    // ---- O += V[ , J:J+128] * P^T  (wave owns 64 channels) ----------------
    const unsigned short* vbp = vbuf + ((size_t)bb * C_ + 64 * wave) * N_ + J;
#pragma unroll
    for (int ct = 0; ct < 4; ++ct) {
      const unsigned short* ar = vbp + (size_t)(16 * ct + ln) * N_ + 8 * lh;
      for (int jj = 0; jj < 128; jj += 32) {
        Frag afr;
        afr.q[0] = *(const uint4*)(ar + jj);
        afr.q[1] = *(const uint4*)(ar + jj + 16);
#pragma unroll
        for (int it = 0; it < IT; ++it) {
          Frag bfr;
          bfr.q[0] = *(const uint4*)&Ps[(it * 16 + ln) * 128 + 8 * lh + jj];
          bfr.q[1] = *(const uint4*)&Ps[(it * 16 + ln) * 128 + 8 * lh + jj + 16];
          oacc[ct][it] = wmma_bf16(afr, bfr, oacc[ct][it]);
        }
      }
    }
    __syncthreads();
  }

  // ---- normalize by l and store transposed [B][N][C] bf16 ------------------
#pragma unroll
  for (int it = 0; it < IT; ++it) {
    const float inv = 1.0f / lrun[it * 16 + ln];
#pragma unroll
    for (int ct = 0; ct < 4; ++ct) {
      Pack8 pk;
#pragma unroll
      for (int r = 0; r < 8; ++r) pk.u[r] = f2bf(oacc[ct][it][r] * inv);
      *(uint4*)(attT + ((size_t)bb * N_ + i0 + it * 16 + ln) * C_ + 64 * wave +
                16 * ct + 8 * lh) = pk.q;
    }
  }
}

// ---------------------------------------------------------------------------
// K4: projection GEMM + bias + residual, f32 output [B][C][N]
// ---------------------------------------------------------------------------
__global__ __launch_bounds__(256) void proj_gemm(
    const unsigned short* __restrict__ Wp, const unsigned short* __restrict__ attT,
    const float* __restrict__ pb, const float* __restrict__ x,
    float* __restrict__ out) {
  const int wave = threadIdx.x >> 5, lane = threadIdx.x & 31;
  const int ln = lane & 15, lh = lane >> 4;
  const int o0 = blockIdx.x * 128 + (wave >> 2) * 64;
  const int n0 = blockIdx.y * 64 + (wave & 3) * 16;
  const int bb = blockIdx.z;

  floatx8 acc[4] = {};
  const unsigned short* bptr = attT + ((size_t)bb * N_ + n0 + ln) * C_ + 8 * lh;
  const unsigned short* aptr = Wp + (size_t)(o0 + ln) * C_ + 8 * lh;

  for (int kc = 0; kc < C_; kc += 32) {
    __builtin_prefetch(bptr + kc + 64, 0, 1);
    Frag bfr;
    bfr.q[0] = *(const uint4*)(bptr + kc);
    bfr.q[1] = *(const uint4*)(bptr + kc + 16);
#pragma unroll
    for (int t = 0; t < 4; ++t) {
      Frag afr;
      const unsigned short* ap = aptr + (size_t)(16 * t) * C_ + kc;
      afr.q[0] = *(const uint4*)(ap);
      afr.q[1] = *(const uint4*)(ap + 16);
      acc[t] = wmma_bf16(afr, bfr, acc[t]);
    }
  }

  const int n = n0 + ln;
#pragma unroll
  for (int t = 0; t < 4; ++t) {
#pragma unroll
    for (int r = 0; r < 8; ++r) {
      const int o = o0 + 16 * t + 8 * lh + r;
      const size_t idx = ((size_t)bb * C_ + o) * N_ + n;
      out[idx] = acc[t][r] + pb[o] + x[idx];
    }
  }
}

// ---------------------------------------------------------------------------
extern "C" void kernel_launch(void* const* d_in, const int* in_sizes, int n_in,
                              void* d_out, int out_size, void* d_ws, size_t ws_size,
                              hipStream_t stream) {
  const float* x    = (const float*)d_in[0];
  const float* gn_g = (const float*)d_in[1];
  const float* gn_b = (const float*)d_in[2];
  const float* q_w  = (const float*)d_in[3];
  const float* q_b  = (const float*)d_in[4];
  const float* k_w  = (const float*)d_in[5];
  const float* k_b  = (const float*)d_in[6];
  const float* v_w  = (const float*)d_in[7];
  const float* v_b  = (const float*)d_in[8];
  const float* p_w  = (const float*)d_in[9];
  const float* p_b  = (const float*)d_in[10];
  float* out = (float*)d_out;

  char* ws = (char*)d_ws;
  size_t off = 0;
  auto carve = [&](size_t bytes) -> void* {
    void* p = ws + off;
    off = (off + bytes + 255) & ~(size_t)255;
    return p;
  };
  const size_t WE = (size_t)C_ * C_;
  const size_t TE = (size_t)B_ * N_ * C_;
  unsigned short* Wqkv = (unsigned short*)carve(3 * WE * 2);
  unsigned short* Wp   = (unsigned short*)carve(WE * 2);
  unsigned short* hT   = (unsigned short*)carve(TE * 2);
  unsigned short* qT   = (unsigned short*)carve(TE * 2);
  unsigned short* kT   = (unsigned short*)carve(TE * 2);
  unsigned short* vbuf = (unsigned short*)carve(TE * 2);
  unsigned short* attT = (unsigned short*)carve(TE * 2);

  const int wgrid = (int)(WE / 256);
  cvt_bf16<<<wgrid, 256, 0, stream>>>(q_w, Wqkv, (int)WE);
  cvt_bf16<<<wgrid, 256, 0, stream>>>(k_w, Wqkv + WE, (int)WE);
  cvt_bf16<<<wgrid, 256, 0, stream>>>(v_w, Wqkv + 2 * WE, (int)WE);
  cvt_bf16<<<wgrid, 256, 0, stream>>>(p_w, Wp, (int)WE);

  gn_kernel<<<B_ * 32, 256, 0, stream>>>(x, gn_g, gn_b, hT);

  qkv_gemm<<<dim3(12, 64, B_), 256, 0, stream>>>(Wqkv, hT, q_b, k_b, v_b, qT, kT, vbuf);

  attn_kernel<<<dim3(N_ / (16 * IT), B_), 256, 0, stream>>>(qT, kT, vbuf, attT);

  proj_gemm<<<dim3(4, 64, B_), 256, 0, stream>>>(Wp, attT, p_b, x, out);
}
